// EGNN_65773129171366
// MI455X (gfx1250) — matrix operand order repaired and checked by reading, not compiled, not run on hardware
//
#include <hip/hip_runtime.h>
#include <hip/hip_bf16.h>

// ---------------------------------------------------------------------------
// EGNN forward + force VJP for MI455X (gfx1250, wave32, WMMA f16->f32).
// Edge/node MLPs fused into tile kernels: 16-row x 128-col tiles, 8 waves per
// 256-thread block, each wave owns one 16-col WMMA N-tile.
// ---------------------------------------------------------------------------

#define NN 20000   // nodes
#define EE 320000  // edges
#define HD 128     // hidden
#define FIN 64
#define DEPTH 4
#define GB 64      // graphs
#define LN_EPS 1e-5f

typedef _Float16 h16_t;
typedef __attribute__((ext_vector_type(16))) _Float16 v16h;
typedef __attribute__((ext_vector_type(8)))  float    v8f;

// ---------------------------------------------------------------------------
// WMMA helper: D[16 x 16] tile at column offset n0 of D = A(16xK) * W(Kx128).
// A: row-major f16 (LDS), row stride K.
// Wt: per-column-contiguous f16 (global): column n lives at Wt + n*K, K contig.
//     (forward weights are stored transposed; backward grad@W^T uses the
//      original row-major weight with K==128, which has the same property.)
// Fragment layouts per CDNA5 ISA 7.12.2 (wave32):
//   A 16x32 f16 : lanes 0-15 row=lane,   elems 0..7 =K k0..k0+7,  8..15=k0+16..23
//                 lanes 16-31 row=lane-16, elems 0..7=k0+8..15, 8..15=k0+24..31
//   B 32x16 f16 : lanes 0-15 col=lane   elems 0..7 =K k0..k0+7,  8..15=k0+16..23
//                 lanes 16-31 col=lane-16 elems 0..7=k0+8..15, 8..15=k0+24..31
//   D 16x16 f32 : vgpr r -> M=r (lanes 0-15) / M=8+r (lanes 16-31), N=lane&15
// ---------------------------------------------------------------------------
__device__ __forceinline__ v8f wmma_rowtile(const h16_t* __restrict__ A, int K,
                                            const h16_t* __restrict__ Wt,
                                            int n0, int lane, v8f acc) {
  const int row = lane & 15;
  const int hi  = lane >> 4;
  const h16_t* arow = A + row * K;
  const h16_t* wcol = Wt + (size_t)(n0 + row) * (size_t)K;
  for (int k0 = 0; k0 < K; k0 += 32) {
    v16h a, b;
    const h16_t* ap = arow + k0 + (hi ? 8 : 0);
    const h16_t* bp = wcol + k0 + (hi ? 8 : 0);
#pragma unroll
    for (int i = 0; i < 8; ++i) { a[i] = ap[i]; a[i + 8] = ap[i + 16]; }
#pragma unroll
    for (int i = 0; i < 8; ++i) { b[i] = bp[i]; b[i + 8] = bp[i + 16]; }
    acc = __builtin_amdgcn_wmma_f32_16x16x32_f16(false, a, false, b,
                                                 (short)0, acc, false, false);
  }
  return acc;
}

__device__ __forceinline__ void store_tile(float* __restrict__ dst /*16xHD LDS*/, int n0, int lane,
                                           v8f d, const float* __restrict__ bias /*global or null*/) {
  const int col = n0 + (lane & 15);
  const int hi  = lane >> 4;
  const float b = bias ? bias[col] : 0.f;
#pragma unroll
  for (int r = 0; r < 8; ++r) dst[(r + (hi ? 8 : 0)) * HD + col] = d[r] + b;
}

// LayerNorm (+gamma/beta) + SiLU over a 16x128 tile held in LDS (z preserved).
// blockDim = 256; wave w handles rows 2w, 2w+1; lane covers 4 columns.
__device__ __forceinline__ void ln_silu_tile(const float* __restrict__ z,
                                             const float* __restrict__ g,
                                             const float* __restrict__ be,
                                             h16_t* __restrict__ out_h,
                                             float* __restrict__ out_f, int tid) {
  const int wave = tid >> 5, lane = tid & 31;
#pragma unroll
  for (int rr = 0; rr < 2; ++rr) {
    const int row = wave * 2 + rr;
    float v[4], s = 0.f, s2 = 0.f;
#pragma unroll
    for (int q = 0; q < 4; ++q) { v[q] = z[row * HD + lane + 32 * q]; s += v[q]; s2 += v[q] * v[q]; }
#pragma unroll
    for (int off = 16; off; off >>= 1) { s += __shfl_xor(s, off, 32); s2 += __shfl_xor(s2, off, 32); }
    const float mean = s * (1.f / HD);
    const float rs = rsqrtf(fmaxf(s2 * (1.f / HD) - mean * mean, 0.f) + LN_EPS);
#pragma unroll
    for (int q = 0; q < 4; ++q) {
      const int c = lane + 32 * q;
      const float y = (v[q] - mean) * rs * g[c] + be[c];
      const float sg = 1.f / (1.f + __expf(-y));
      const float o = y * sg;
      out_h[row * HD + c] = (h16_t)o;
      if (out_f) out_f[row * HD + c] = o;
    }
  }
}

// Backward through SiLU(LN(z)): gio holds upstream grad, replaced by dL/dz.
__device__ __forceinline__ void ln_silu_bwd_tile(const float* __restrict__ z,
                                                 float* __restrict__ gio,
                                                 const float* __restrict__ g,
                                                 const float* __restrict__ be, int tid) {
  const int wave = tid >> 5, lane = tid & 31;
#pragma unroll
  for (int rr = 0; rr < 2; ++rr) {
    const int row = wave * 2 + rr;
    float v[4], s = 0.f, s2 = 0.f;
#pragma unroll
    for (int q = 0; q < 4; ++q) { v[q] = z[row * HD + lane + 32 * q]; s += v[q]; s2 += v[q] * v[q]; }
#pragma unroll
    for (int off = 16; off; off >>= 1) { s += __shfl_xor(s, off, 32); s2 += __shfl_xor(s2, off, 32); }
    const float mean = s * (1.f / HD);
    const float rs = rsqrtf(fmaxf(s2 * (1.f / HD) - mean * mean, 0.f) + LN_EPS);
    float t[4], xh[4], ts = 0.f, txs = 0.f;
#pragma unroll
    for (int q = 0; q < 4; ++q) {
      const int c = lane + 32 * q;
      xh[q] = (v[q] - mean) * rs;
      const float y = xh[q] * g[c] + be[c];
      const float sg = 1.f / (1.f + __expf(-y));
      const float dsilu = sg * (1.f + y * (1.f - sg));
      t[q] = gio[row * HD + c] * dsilu * g[c];
      ts += t[q]; txs += t[q] * xh[q];
    }
#pragma unroll
    for (int off = 16; off; off >>= 1) { ts += __shfl_xor(ts, off, 32); txs += __shfl_xor(txs, off, 32); }
    const float m1 = ts * (1.f / HD), m2 = txs * (1.f / HD);
#pragma unroll
    for (int q = 0; q < 4; ++q) {
      const int c = lane + 32 * q;
      gio[row * HD + c] = rs * (t[q] - m1 - xh[q] * m2);
    }
  }
}

// ------------------------------ utility kernels -----------------------------
__global__ void k_zero(float* __restrict__ p, long n) {
  for (long t = blockIdx.x * 256L + threadIdx.x; t < n; t += (long)gridDim.x * 256L) p[t] = 0.f;
}
__global__ void k_cvt_h(h16_t* __restrict__ d, const float* __restrict__ s, long n) {
  for (long t = blockIdx.x * 256L + threadIdx.x; t < n; t += (long)gridDim.x * 256L) d[t] = (h16_t)s[t];
}
// dst[c*K + k] = src[k*128 + c]  (store forward weights column-major f16)
__global__ void k_transpose_h(h16_t* __restrict__ d, const float* __restrict__ s, int K) {
  const long n = (long)K * HD;
  for (long t = blockIdx.x * 256L + threadIdx.x; t < n; t += (long)gridDim.x * 256L) {
    const int k = (int)(t >> 7), c = (int)(t & 127);
    d[(size_t)c * K + k] = (h16_t)s[(size_t)k * HD + c];
  }
}

// distance + unit vector: dist = ||pos_i - off@U - pos_j||
__global__ void k_dist(const float* __restrict__ pos, const int* __restrict__ idx_j,
                       const int* __restrict__ idx_i, const float* __restrict__ off,
                       const float* __restrict__ U, float* __restrict__ dist,
                       float* __restrict__ dvec) {
  for (long e = blockIdx.x * 256L + threadIdx.x; e < EE; e += (long)gridDim.x * 256L) {
    const int i = idx_i[e], j = idx_j[e];
    const float o0 = off[e * 3 + 0], o1 = off[e * 3 + 1], o2 = off[e * 3 + 2];
    float d[3], dd = 0.f;
#pragma unroll
    for (int k = 0; k < 3; ++k) {
      const float corr = pos[(size_t)i * 3 + k] - (o0 * U[0 * 3 + k] + o1 * U[1 * 3 + k] + o2 * U[2 * 3 + k]);
      d[k] = corr - pos[(size_t)j * 3 + k];
      dd += d[k] * d[k];
    }
    const float r = sqrtf(dd);
    const float inv = (r > 0.f) ? 1.f / r : 0.f;
    dist[e] = r;
#pragma unroll
    for (int k = 0; k < 3; ++k) dvec[e * 3 + k] = d[k] * inv;
  }
}

// ------------------------------ embedding -----------------------------------
__global__ void k_embed(const float* __restrict__ x, const h16_t* __restrict__ wt /*128 cols x 64*/,
                        const float* __restrict__ bias,
                        float* __restrict__ h0, h16_t* __restrict__ h16) {
  __shared__ h16_t A[16 * FIN];
  const int tid = threadIdx.x, lane = tid & 31, wave = tid >> 5;
  for (int tile = blockIdx.x; tile < NN / 16; tile += gridDim.x) {
    const int node0 = tile * 16;
    for (int t = tid; t < 16 * FIN; t += 256) {
      const int r = t >> 6, c = t & 63;
      A[t] = (h16_t)x[(size_t)(node0 + r) * FIN + c];
    }
    __syncthreads();
    v8f acc = {0.f, 0.f, 0.f, 0.f, 0.f, 0.f, 0.f, 0.f};
    acc = wmma_rowtile(A, FIN, wt, wave * 16, lane, acc);
    const int col = wave * 16 + (lane & 15);
    const int hi = lane >> 4;
    const float bc = bias[col];       // hoisted: loop-invariant
#pragma unroll
    for (int r = 0; r < 8; ++r) {
      const int m = r + (hi ? 8 : 0);
      const float v = acc[r] + bc;
      h0[(size_t)(node0 + m) * HD + col] = v;
      h16[(size_t)(node0 + m) * HD + col] = (h16_t)v;
    }
    __syncthreads();
  }
}

// ------------------------------ edge message forward ------------------------
__global__ void k_edge_fwd(const h16_t* __restrict__ h16, const float* __restrict__ dist,
                           const int* __restrict__ idx_j, const int* __restrict__ idx_i,
                           const h16_t* __restrict__ wt1 /*col-major 256*/,
                           const h16_t* __restrict__ rm1 /*row-major 257x128*/,
                           const h16_t* __restrict__ wt2 /*col-major 128*/,
                           const float* __restrict__ b1, const float* __restrict__ g1,
                           const float* __restrict__ be1,
                           const float* __restrict__ b2, const float* __restrict__ g2,
                           const float* __restrict__ be2,
                           float* __restrict__ agg) {
  __shared__ h16_t A[16 * 256];
  __shared__ h16_t A2[16 * HD];
  __shared__ float z[16 * HD];
  __shared__ float mout[16 * HD];
  __shared__ int si[16], sj[16];
  __shared__ float sd[16];
  const int tid = threadIdx.x, lane = tid & 31, wave = tid >> 5;
  const h16_t* w1last = rm1 + 256 * HD;
  for (int tile = blockIdx.x; tile < EE / 16; tile += gridDim.x) {
    const int e0 = tile * 16;
    if (tid < 16) { si[tid] = idx_i[e0 + tid]; sj[tid] = idx_j[e0 + tid]; sd[tid] = dist[e0 + tid]; }
    // prefetch next tile's indices / dist (global_prefetch_b8)
    {
      const int en = e0 + (int)gridDim.x * 16;
      if (en < EE && tid < 16) {
        __builtin_prefetch(&idx_i[en + tid], 0, 1);
        __builtin_prefetch(&idx_j[en + tid], 0, 1);
        __builtin_prefetch(&dist[en + tid], 0, 1);
      }
    }
    __syncthreads();
    // gather [h_i | h_j] as u32 pairs
    for (int t = tid; t < 2048; t += 256) {
      const int r = t >> 7, q = t & 127;
      const unsigned int* src = (q < 64)
          ? (const unsigned int*)(h16 + (size_t)si[r] * HD) + q
          : (const unsigned int*)(h16 + (size_t)sj[r] * HD) + (q - 64);
      ((unsigned int*)A)[r * 128 + q] = *src;
    }
    __syncthreads();
    const int n0 = wave * 16;
    v8f acc = {0.f, 0.f, 0.f, 0.f, 0.f, 0.f, 0.f, 0.f};
    acc = wmma_rowtile(A, 256, wt1, n0, lane, acc);
    store_tile(z, n0, lane, acc, b1);
    {   // dist column (row 256 of msg_w1)
      const int col = n0 + (lane & 15);
      const int hi = lane >> 4;
      const float wl = (float)w1last[col];
#pragma unroll
      for (int r = 0; r < 8; ++r) {
        const int m = r + (hi ? 8 : 0);
        z[m * HD + col] += sd[m] * wl;
      }
    }
    __syncthreads();
    ln_silu_tile(z, g1, be1, A2, nullptr, tid);
    __syncthreads();
    v8f acc2 = {0.f, 0.f, 0.f, 0.f, 0.f, 0.f, 0.f, 0.f};
    acc2 = wmma_rowtile(A2, HD, wt2, n0, lane, acc2);
    store_tile(z, n0, lane, acc2, b2);
    __syncthreads();
    ln_silu_tile(z, g2, be2, A2, mout, tid);
    __syncthreads();
    for (int t = tid; t < 16 * HD; t += 256) {
      const int r = t >> 7, c = t & 127;
      atomicAdd(&agg[(size_t)si[r] * HD + c], mout[t]);
    }
    __syncthreads();
  }
}

// ------------------------------ node update forward -------------------------
__global__ void k_node_fwd(const float* __restrict__ hl, const float* __restrict__ agg,
                           const h16_t* __restrict__ wt1, const h16_t* __restrict__ wt2,
                           const float* __restrict__ b1, const float* __restrict__ g1,
                           const float* __restrict__ be1,
                           const float* __restrict__ b2, const float* __restrict__ g2,
                           const float* __restrict__ be2,
                           float* __restrict__ hout, h16_t* __restrict__ h16o) {
  __shared__ h16_t A[16 * 256];
  __shared__ h16_t A2[16 * HD];
  __shared__ float z[16 * HD];
  __shared__ float mout[16 * HD];
  const int tid = threadIdx.x, lane = tid & 31, wave = tid >> 5;
  for (int tile = blockIdx.x; tile < NN / 16; tile += gridDim.x) {
    const int node0 = tile * 16;
    for (int t = tid; t < 16 * 256; t += 256) {
      const int r = t >> 8, c = t & 255;
      const float v = (c < HD) ? hl[(size_t)(node0 + r) * HD + c]
                               : agg[(size_t)(node0 + r) * HD + (c - HD)];
      A[t] = (h16_t)v;
    }
    __syncthreads();
    const int n0 = wave * 16;
    v8f acc = {0.f, 0.f, 0.f, 0.f, 0.f, 0.f, 0.f, 0.f};
    acc = wmma_rowtile(A, 256, wt1, n0, lane, acc);
    store_tile(z, n0, lane, acc, b1);
    __syncthreads();
    ln_silu_tile(z, g1, be1, A2, nullptr, tid);
    __syncthreads();
    v8f acc2 = {0.f, 0.f, 0.f, 0.f, 0.f, 0.f, 0.f, 0.f};
    acc2 = wmma_rowtile(A2, HD, wt2, n0, lane, acc2);
    store_tile(z, n0, lane, acc2, b2);
    __syncthreads();
    ln_silu_tile(z, g2, be2, A2, mout, tid);
    __syncthreads();
    for (int t = tid; t < 16 * HD; t += 256) {
      const int r = t >> 7, c = t & 127;
      const float v = hl[(size_t)(node0 + r) * HD + c] + mout[t];
      hout[(size_t)(node0 + r) * HD + c] = v;
      h16o[(size_t)(node0 + r) * HD + c] = (h16_t)v;
    }
    __syncthreads();
  }
}

// ------------------------------ pooling + head ------------------------------
__global__ void k_pool(const float* __restrict__ h4, const int* __restrict__ batch,
                       float* __restrict__ pooled) {
  for (long t = blockIdx.x * 256L + threadIdx.x; t < (long)NN * HD; t += (long)gridDim.x * 256L) {
    const int n = (int)(t >> 7), c = (int)(t & 127);
    atomicAdd(&pooled[(size_t)batch[n] * HD + c], h4[t]);
  }
}

__global__ void k_head(const float* __restrict__ pooled, const float* __restrict__ pw1,
                       const float* __restrict__ pb1, const float* __restrict__ pw2,
                       const float* __restrict__ pb2,
                       float* __restrict__ energy, float* __restrict__ gpool) {
  __shared__ float zs[GB * HD];
  const int tid = threadIdx.x;
  for (int t = tid; t < GB * HD; t += 256) {
    const int g = t >> 7, j = t & 127;
    float s = pb1[j];
    for (int k = 0; k < HD; ++k) s += pooled[(size_t)g * HD + k] * pw1[(size_t)k * HD + j];
    zs[t] = s;
  }
  __syncthreads();
  if (tid < GB) {
    float e = pb2[0];
    for (int j = 0; j < HD; ++j) e += fmaxf(zs[tid * HD + j], 0.f) * pw2[j];
    energy[tid] = e;
  }
  for (int t = tid; t < GB * HD; t += 256) {
    const int g = t >> 7, k = t & 127;
    float s = 0.f;
    for (int j = 0; j < HD; ++j)
      s += (zs[g * HD + j] > 0.f) ? pw2[j] * pw1[(size_t)k * HD + j] : 0.f;
    gpool[t] = s;
  }
}

__global__ void k_bcast(const float* __restrict__ gpool, const int* __restrict__ batch,
                        float* __restrict__ gA) {
  for (long t = blockIdx.x * 256L + threadIdx.x; t < (long)NN * HD; t += (long)gridDim.x * 256L) {
    const int n = (int)(t >> 7), c = (int)(t & 127);
    gA[t] = gpool[(size_t)batch[n] * HD + c];
  }
}

// ------------------------------ node update backward ------------------------
__global__ void k_node_bwd(const float* __restrict__ hl, const float* __restrict__ agg,
                           const h16_t* __restrict__ wt1, const h16_t* __restrict__ wt2, // fwd recompute
                           const h16_t* __restrict__ rm1, const h16_t* __restrict__ rm2, // grad @ W^T
                           const float* __restrict__ b1, const float* __restrict__ g1,
                           const float* __restrict__ be1,
                           const float* __restrict__ b2, const float* __restrict__ g2,
                           const float* __restrict__ be2,
                           const float* __restrict__ gin, float* __restrict__ gout,
                           float* __restrict__ gagg) {
  __shared__ h16_t A[16 * 256];
  __shared__ h16_t A2[16 * HD];
  __shared__ float z1[16 * HD];
  __shared__ float z2[16 * HD];
  __shared__ float gb[16 * HD];
  const int tid = threadIdx.x, lane = tid & 31, wave = tid >> 5;
  for (int tile = blockIdx.x; tile < NN / 16; tile += gridDim.x) {
    const int node0 = tile * 16;
    for (int t = tid; t < 16 * 256; t += 256) {
      const int r = t >> 8, c = t & 255;
      const float v = (c < HD) ? hl[(size_t)(node0 + r) * HD + c]
                               : agg[(size_t)(node0 + r) * HD + (c - HD)];
      A[t] = (h16_t)v;
    }
    __syncthreads();
    const int n0 = wave * 16;
    v8f a1 = {0.f, 0.f, 0.f, 0.f, 0.f, 0.f, 0.f, 0.f};
    a1 = wmma_rowtile(A, 256, wt1, n0, lane, a1);
    store_tile(z1, n0, lane, a1, b1);
    __syncthreads();
    ln_silu_tile(z1, g1, be1, A2, nullptr, tid);
    __syncthreads();
    v8f a2 = {0.f, 0.f, 0.f, 0.f, 0.f, 0.f, 0.f, 0.f};
    a2 = wmma_rowtile(A2, HD, wt2, n0, lane, a2);
    store_tile(z2, n0, lane, a2, b2);
    __syncthreads();
    for (int t = tid; t < 16 * HD; t += 256)
      gb[t] = gin[(size_t)(node0 + (t >> 7)) * HD + (t & 127)];
    __syncthreads();
    ln_silu_bwd_tile(z2, gb, g2, be2, tid);   // gb = dL/dz2
    __syncthreads();
    for (int t = tid; t < 16 * HD; t += 256) A2[t] = (h16_t)gb[t];
    __syncthreads();
    v8f a3 = {0.f, 0.f, 0.f, 0.f, 0.f, 0.f, 0.f, 0.f};
    a3 = wmma_rowtile(A2, HD, rm2, n0, lane, a3);  // dL/du1 = gz2 @ W2^T
    store_tile(gb, n0, lane, a3, nullptr);
    __syncthreads();
    ln_silu_bwd_tile(z1, gb, g1, be1, tid);   // gb = dL/dz1
    __syncthreads();
    for (int t = tid; t < 16 * HD; t += 256) A2[t] = (h16_t)gb[t];
    __syncthreads();
#pragma unroll
    for (int half2 = 0; half2 < 2; ++half2) {
      const int n0b = wave * 16 + half2 * HD;
      v8f a4 = {0.f, 0.f, 0.f, 0.f, 0.f, 0.f, 0.f, 0.f};
      a4 = wmma_rowtile(A2, HD, rm1, n0b, lane, a4);  // dL/d[h|agg] = gz1 @ W1^T
      const int col = n0b + (lane & 15);
      const int hi = lane >> 4;
#pragma unroll
      for (int r = 0; r < 8; ++r) {
        const int m = r + (hi ? 8 : 0);
        const size_t node = (size_t)(node0 + m);
        if (half2 == 0)
          gout[node * HD + col] = gin[node * HD + col] + a4[r];  // residual + update path
        else
          gagg[node * HD + (col - HD)] = a4[r];
      }
    }
    __syncthreads();
  }
}

// ------------------------------ edge message backward -----------------------
__global__ void k_edge_bwd(const h16_t* __restrict__ h16, const float* __restrict__ dist,
                           const int* __restrict__ idx_j, const int* __restrict__ idx_i,
                           const h16_t* __restrict__ wt1, const h16_t* __restrict__ wt2, // fwd recompute
                           const h16_t* __restrict__ rm1 /*257x128*/, const h16_t* __restrict__ rm2,
                           const float* __restrict__ b1, const float* __restrict__ g1,
                           const float* __restrict__ be1,
                           const float* __restrict__ b2, const float* __restrict__ g2,
                           const float* __restrict__ be2,
                           const float* __restrict__ gagg, float* __restrict__ gh,
                           float* __restrict__ gdist) {
  __shared__ h16_t A[16 * 256];
  __shared__ h16_t A2[16 * HD];
  __shared__ float z1[16 * HD];
  __shared__ float z2[16 * HD];
  __shared__ float gb[16 * HD];
  __shared__ int si[16], sj[16];
  __shared__ float sd[16];
  const int tid = threadIdx.x, lane = tid & 31, wave = tid >> 5;
  const h16_t* w1last = rm1 + 256 * HD;
  for (int tile = blockIdx.x; tile < EE / 16; tile += gridDim.x) {
    const int e0 = tile * 16;
    if (tid < 16) { si[tid] = idx_i[e0 + tid]; sj[tid] = idx_j[e0 + tid]; sd[tid] = dist[e0 + tid]; }
    {
      const int en = e0 + (int)gridDim.x * 16;
      if (en < EE && tid < 16) {
        __builtin_prefetch(&idx_i[en + tid], 0, 1);
        __builtin_prefetch(&idx_j[en + tid], 0, 1);
        __builtin_prefetch(&dist[en + tid], 0, 1);
      }
    }
    __syncthreads();
    for (int t = tid; t < 2048; t += 256) {
      const int r = t >> 7, q = t & 127;
      const unsigned int* src = (q < 64)
          ? (const unsigned int*)(h16 + (size_t)si[r] * HD) + q
          : (const unsigned int*)(h16 + (size_t)sj[r] * HD) + (q - 64);
      ((unsigned int*)A)[r * 128 + q] = *src;
    }
    __syncthreads();
    const int n0 = wave * 16;
    v8f a1 = {0.f, 0.f, 0.f, 0.f, 0.f, 0.f, 0.f, 0.f};
    a1 = wmma_rowtile(A, 256, wt1, n0, lane, a1);
    store_tile(z1, n0, lane, a1, b1);
    {
      const int col = n0 + (lane & 15);
      const int hi = lane >> 4;
      const float wl = (float)w1last[col];
#pragma unroll
      for (int r = 0; r < 8; ++r) {
        const int m = r + (hi ? 8 : 0);
        z1[m * HD + col] += sd[m] * wl;
      }
    }
    __syncthreads();
    ln_silu_tile(z1, g1, be1, A2, nullptr, tid);
    __syncthreads();
    v8f a2 = {0.f, 0.f, 0.f, 0.f, 0.f, 0.f, 0.f, 0.f};
    a2 = wmma_rowtile(A2, HD, wt2, n0, lane, a2);
    store_tile(z2, n0, lane, a2, b2);
    __syncthreads();
    for (int t = tid; t < 16 * HD; t += 256)     // dL/dm = gagg[idx_i]
      gb[t] = gagg[(size_t)si[t >> 7] * HD + (t & 127)];
    __syncthreads();
    ln_silu_bwd_tile(z2, gb, g2, be2, tid);      // gb = dL/dz2
    __syncthreads();
    for (int t = tid; t < 16 * HD; t += 256) A2[t] = (h16_t)gb[t];
    __syncthreads();
    v8f a3 = {0.f, 0.f, 0.f, 0.f, 0.f, 0.f, 0.f, 0.f};
    a3 = wmma_rowtile(A2, HD, rm2, n0, lane, a3);  // dL/dm1 = gz2 @ W2^T
    store_tile(gb, n0, lane, a3, nullptr);
    __syncthreads();
    ln_silu_bwd_tile(z1, gb, g1, be1, tid);      // gb = dL/dz1
    __syncthreads();
    // dL/ddist[e] += gz1[e,:] . W1[256,:]
#pragma unroll
    for (int rr = 0; rr < 2; ++rr) {
      const int row = wave * 2 + rr;
      float s = 0.f;
#pragma unroll
      for (int q = 0; q < 4; ++q) {
        const int c = lane + 32 * q;
        s += gb[row * HD + c] * (float)w1last[c];
      }
#pragma unroll
      for (int off = 16; off; off >>= 1) s += __shfl_xor(s, off, 32);
      if (lane == 0) gdist[e0 + row] += s;
    }
    for (int t = tid; t < 16 * HD; t += 256) A2[t] = (h16_t)gb[t];
    __syncthreads();
#pragma unroll
    for (int half2 = 0; half2 < 2; ++half2) {
      const int n0b = wave * 16 + half2 * HD;
      v8f a4 = {0.f, 0.f, 0.f, 0.f, 0.f, 0.f, 0.f, 0.f};
      a4 = wmma_rowtile(A2, HD, rm1, n0b, lane, a4);  // dL/d[h_i|h_j]
      const int col = n0b + (lane & 15);
      const int hi = lane >> 4;
#pragma unroll
      for (int r = 0; r < 8; ++r) {
        const int m = r + (hi ? 8 : 0);
        if (half2 == 0)
          atomicAdd(&gh[(size_t)si[m] * HD + col], a4[r]);
        else
          atomicAdd(&gh[(size_t)sj[m] * HD + (col - HD)], a4[r]);
      }
    }
    __syncthreads();
  }
}

// ------------------------------ force ---------------------------------------
__global__ void k_force(const float* __restrict__ gdist, const float* __restrict__ dvec,
                        const int* __restrict__ idx_j, const int* __restrict__ idx_i,
                        float* __restrict__ force) {
  for (long e = blockIdx.x * 256L + threadIdx.x; e < EE; e += (long)gridDim.x * 256L) {
    const float gd = gdist[e];
    const int i = idx_i[e], j = idx_j[e];
#pragma unroll
    for (int k = 0; k < 3; ++k) {
      const float f = gd * dvec[e * 3 + k];
      atomicAdd(&force[(size_t)i * 3 + k], -f);  // force = -dE/dpos
      atomicAdd(&force[(size_t)j * 3 + k],  f);
    }
  }
}

// ============================================================================
extern "C" void kernel_launch(void* const* d_in, const int* in_sizes, int n_in,
                              void* d_out, int out_size, void* d_ws, size_t ws_size,
                              hipStream_t stream) {
  (void)in_sizes; (void)n_in; (void)out_size; (void)ws_size;
  const float* x       = (const float*)d_in[0];
  const float* pos     = (const float*)d_in[1];
  const int*   eidx    = (const int*)d_in[2];
  const float* celloff = (const float*)d_in[3];
  const float* ucell   = (const float*)d_in[4];
  const int*   batch   = (const int*)d_in[5];
  const float* embW    = (const float*)d_in[6];
  const float* embb    = (const float*)d_in[7];
  const float* msgw1   = (const float*)d_in[8];
  const float* msgb1   = (const float*)d_in[9];
  const float* msgg1   = (const float*)d_in[10];
  const float* msgbe1  = (const float*)d_in[11];
  const float* msgw2   = (const float*)d_in[12];
  const float* msgb2   = (const float*)d_in[13];
  const float* msgg2   = (const float*)d_in[14];
  const float* msgbe2  = (const float*)d_in[15];
  const float* updw1   = (const float*)d_in[16];
  const float* updb1   = (const float*)d_in[17];
  const float* updg1   = (const float*)d_in[18];
  const float* updbe1  = (const float*)d_in[19];
  const float* updw2   = (const float*)d_in[20];
  const float* updb2   = (const float*)d_in[21];
  const float* updg2   = (const float*)d_in[22];
  const float* updbe2  = (const float*)d_in[23];
  const float* pw1     = (const float*)d_in[24];
  const float* pb1     = (const float*)d_in[25];
  const float* pw2     = (const float*)d_in[26];
  const float* pb2     = (const float*)d_in[27];
  const int* idx_j = eidx;        // source (x_j)
  const int* idx_i = eidx + EE;   // target / aggregation index
  float* energy = (float*)d_out;
  float* force  = (float*)d_out + GB;

  // ---- carve workspace ----
  char* wsp = (char*)d_ws;
  auto carve = [&](size_t bytes) { char* p = wsp; wsp += (bytes + 255) & ~(size_t)255; return p; };
  h16_t* wt_emb = (h16_t*)carve((size_t)FIN * HD * 2);
  h16_t* wt_m1  = (h16_t*)carve((size_t)DEPTH * 256 * HD * 2);
  h16_t* wt_m2  = (h16_t*)carve((size_t)DEPTH * HD * HD * 2);
  h16_t* wt_u1  = (h16_t*)carve((size_t)DEPTH * 256 * HD * 2);
  h16_t* wt_u2  = (h16_t*)carve((size_t)DEPTH * HD * HD * 2);
  h16_t* rm_m1  = (h16_t*)carve((size_t)DEPTH * 257 * HD * 2);
  h16_t* rm_m2  = (h16_t*)carve((size_t)DEPTH * HD * HD * 2);
  h16_t* rm_u1  = (h16_t*)carve((size_t)DEPTH * 256 * HD * 2);
  h16_t* rm_u2  = (h16_t*)carve((size_t)DEPTH * HD * HD * 2);
  h16_t* h16    = (h16_t*)carve((size_t)NN * HD * 2);
  float* hchk   = (float*)carve((size_t)(DEPTH + 1) * NN * HD * 4);
  float* agg    = (float*)carve((size_t)DEPTH * NN * HD * 4);
  float* dist   = (float*)carve((size_t)EE * 4);
  float* dvec   = (float*)carve((size_t)EE * 3 * 4);
  float* pooled = (float*)carve((size_t)GB * HD * 4);
  float* gpool  = (float*)carve((size_t)GB * HD * 4);
  float* gA     = (float*)carve((size_t)NN * HD * 4);
  float* gBuf   = (float*)carve((size_t)NN * HD * 4);
  float* gagg   = (float*)carve((size_t)NN * HD * 4);
  float* gdist  = (float*)carve((size_t)EE * 4);

  const int NT = NN / 16;   // 1250 node tiles
  const int ET = EE / 16;   // 20000 edge tiles
  const long NH = (long)NN * HD;

  // ---- weight prep (f16 copies: forward col-major, backward row-major) ----
  k_transpose_h<<<dim3(32), dim3(256), 0, stream>>>(wt_emb, embW, FIN);
  for (int l = 0; l < DEPTH; ++l) {
    k_transpose_h<<<dim3(128), dim3(256), 0, stream>>>(wt_m1 + (size_t)l * 256 * HD, msgw1 + (size_t)l * 257 * HD, 256);
    k_transpose_h<<<dim3(64),  dim3(256), 0, stream>>>(wt_m2 + (size_t)l * HD * HD,  msgw2 + (size_t)l * HD * HD,  HD);
    k_transpose_h<<<dim3(128), dim3(256), 0, stream>>>(wt_u1 + (size_t)l * 256 * HD, updw1 + (size_t)l * 256 * HD, 256);
    k_transpose_h<<<dim3(64),  dim3(256), 0, stream>>>(wt_u2 + (size_t)l * HD * HD,  updw2 + (size_t)l * HD * HD,  HD);
  }
  k_cvt_h<<<dim3(520), dim3(256), 0, stream>>>(rm_m1, msgw1, (long)DEPTH * 257 * HD);
  k_cvt_h<<<dim3(256), dim3(256), 0, stream>>>(rm_m2, msgw2, (long)DEPTH * HD * HD);
  k_cvt_h<<<dim3(512), dim3(256), 0, stream>>>(rm_u1, updw1, (long)DEPTH * 256 * HD);
  k_cvt_h<<<dim3(256), dim3(256), 0, stream>>>(rm_u2, updw2, (long)DEPTH * HD * HD);

  // ---- forward ----
  k_dist<<<dim3(1024), dim3(256), 0, stream>>>(pos, idx_j, idx_i, celloff, ucell, dist, dvec);
  k_embed<<<dim3(NT), dim3(256), 0, stream>>>(x, wt_emb, embb, hchk, h16);
  for (int l = 0; l < DEPTH; ++l) {
    float* aggl = agg + (size_t)l * NH;
    k_zero<<<dim3(4096), dim3(256), 0, stream>>>(aggl, NH);
    k_edge_fwd<<<dim3(ET), dim3(256), 0, stream>>>(
        h16, dist, idx_j, idx_i,
        wt_m1 + (size_t)l * 256 * HD, rm_m1 + (size_t)l * 257 * HD, wt_m2 + (size_t)l * HD * HD,
        msgb1 + l * HD, msgg1 + l * HD, msgbe1 + l * HD,
        msgb2 + l * HD, msgg2 + l * HD, msgbe2 + l * HD, aggl);
    k_node_fwd<<<dim3(NT), dim3(256), 0, stream>>>(
        hchk + (size_t)l * NH, aggl,
        wt_u1 + (size_t)l * 256 * HD, wt_u2 + (size_t)l * HD * HD,
        updb1 + l * HD, updg1 + l * HD, updbe1 + l * HD,
        updb2 + l * HD, updg2 + l * HD, updbe2 + l * HD,
        hchk + (size_t)(l + 1) * NH, h16);
  }
  k_zero<<<dim3(32), dim3(256), 0, stream>>>(pooled, (long)GB * HD);
  k_pool<<<dim3(4096), dim3(256), 0, stream>>>(hchk + (size_t)DEPTH * NH, batch, pooled);
  k_head<<<dim3(1), dim3(256), 0, stream>>>(pooled, pw1, pb1, pw2, pb2, energy, gpool);

  // ---- backward ----
  k_bcast<<<dim3(4096), dim3(256), 0, stream>>>(gpool, batch, gA);
  k_zero<<<dim3(1024), dim3(256), 0, stream>>>(gdist, EE);
  float* gin = gA;
  float* gout = gBuf;
  for (int l = DEPTH - 1; l >= 0; --l) {
    float* aggl = agg + (size_t)l * NH;
    k_cvt_h<<<dim3(4096), dim3(256), 0, stream>>>(h16, hchk + (size_t)l * NH, NH);
    k_node_bwd<<<dim3(NT), dim3(256), 0, stream>>>(
        hchk + (size_t)l * NH, aggl,
        wt_u1 + (size_t)l * 256 * HD, wt_u2 + (size_t)l * HD * HD,
        rm_u1 + (size_t)l * 256 * HD, rm_u2 + (size_t)l * HD * HD,
        updb1 + l * HD, updg1 + l * HD, updbe1 + l * HD,
        updb2 + l * HD, updg2 + l * HD, updbe2 + l * HD,
        gin, gout, gagg);
    k_edge_bwd<<<dim3(ET), dim3(256), 0, stream>>>(
        h16, dist, idx_j, idx_i,
        wt_m1 + (size_t)l * 256 * HD, wt_m2 + (size_t)l * HD * HD,
        rm_m1 + (size_t)l * 257 * HD, rm_m2 + (size_t)l * HD * HD,
        msgb1 + l * HD, msgg1 + l * HD, msgbe1 + l * HD,
        msgb2 + l * HD, msgg2 + l * HD, msgbe2 + l * HD,
        gagg, gout, gdist);
    float* t = gin; gin = gout; gout = t;
  }
  k_zero<<<dim3(1024), dim3(256), 0, stream>>>(force, (long)NN * 3);
  k_force<<<dim3(1024), dim3(256), 0, stream>>>(gdist, dvec, idx_j, idx_i, force);
}
